// ContextualAttention_13434657702224
// MI455X (gfx1250) — matrix operand order repaired and checked by reading, not compiled, not run on hardware
//
#include <hip/hip_runtime.h>

// ---------------- types for CDNA5 WMMA ----------------
typedef __bf16 bf16;
typedef bf16  v8bf  __attribute__((ext_vector_type(8)));
typedef bf16  v16bf __attribute__((ext_vector_type(16)));
typedef float v8f   __attribute__((ext_vector_type(8)));

// ---------------- problem constants ----------------
#define BATCH 4
#define HH 96
#define CC 128
#define hh 48
#define LL 2304          // hh*hh
#define EE 1152          // 3*3*CC

// ---------------- kernel 1: bilinear half-resize == 2x2 avg pool ----------------
__global__ void k_pool(const float* __restrict__ x, const float* __restrict__ mask,
                       float* __restrict__ f, float* __restrict__ m) {
    int idx = blockIdx.x * blockDim.x + threadIdx.x;
    const int totalF = hh * hh * CC;
    if (idx < totalF) {
        int c = idx & (CC - 1);
        int j = (idx >> 7) % hh;
        int i = idx / (hh * CC);
        const float* x0 = x + ((size_t)(2 * i) * HH + 2 * j) * CC + c;
        f[idx] = 0.25f * (x0[0] + x0[CC] + x0[(size_t)HH * CC] + x0[(size_t)HH * CC + CC]);
    }
    if (idx < hh * hh) {
        int j = idx % hh, i = idx / hh;
        const float* m0 = mask + ((size_t)(2 * i) * HH + 2 * j);
        m[idx] = 0.25f * (m0[0] + m0[1] + m0[HH] + m0[HH + 1]);
    }
}

// ---------------- kernel 2: build patch matrix P (bf16), inv_norm, valid mask ----------------
__global__ __launch_bounds__(256) void k_patch(const float* __restrict__ f,
                                               const float* __restrict__ m,
                                               bf16* __restrict__ P,
                                               float* __restrict__ invn,
                                               float* __restrict__ mm) {
    __shared__ float red[256];
    int l = blockIdx.x;          // patch index 0..L-1
    int i = l / hh, j = l % hh;
    float ssum = 0.f;
    for (int k = threadIdx.x; k < EE; k += 256) {
        int a = k / (3 * CC);
        int b = (k / CC) % 3;
        int c = k & (CC - 1);
        int y = i + a - 1, xq = j + b - 1;
        float v = 0.f;
        if (y >= 0 && y < hh && xq >= 0 && xq < hh) v = f[((size_t)y * hh + xq) * CC + c];
        P[(size_t)l * EE + k] = (bf16)v;
        ssum += v * v;
    }
    red[threadIdx.x] = ssum;
    __syncthreads();
    for (int s = 128; s > 0; s >>= 1) {
        if (threadIdx.x < s) red[threadIdx.x] += red[threadIdx.x + s];
        __syncthreads();
    }
    if (threadIdx.x == 0) {
        float nrm = sqrtf(red[0]);
        invn[l] = 1.f / fmaxf(nrm, 1e-4f);
        float msum = 0.f;
        for (int a = 0; a < 3; a++)
            for (int b = 0; b < 3; b++) {
                int y = i + a - 1, xq = j + b - 1;
                if (y >= 0 && y < hh && xq >= 0 && xq < hh) msum += m[y * hh + xq];
            }
        mm[l] = (msum == 0.f) ? 1.f : 0.f;
    }
}

// ---------------- kernel 3: transposed raw-patch matrix XabT[ab][c][l] (bf16) ----------------
__global__ void k_xabt(const float* __restrict__ x, bf16* __restrict__ XabT) {
    int idx = blockIdx.x * blockDim.x + threadIdx.x;   // 9*CC*LL
    if (idx >= 9 * CC * LL) return;
    int l  = idx % LL;
    int c  = (idx / LL) & (CC - 1);
    int ab = idx / (LL * CC);
    int a = ab / 3, b = ab % 3;
    int i = l / hh, j = l % hh;
    int p = 2 * i + a, q = 2 * j + b;                  // SAME pad: top/left 0, bottom/right 1
    float v = (p < HH && q < HH) ? x[((size_t)p * HH + q) * CC + c] : 0.f;
    XabT[idx] = (bf16)v;
}

// ---------------- fragment loaders (documented CDNA5 16-bit A/B layouts) ----------------
__device__ inline v16bf load_afrag(const bf16* __restrict__ row, int k0, int hi) {
    // A 16x32: lanes 0-15 hold K = k0+{0..7,16..23}; lanes 16-31 hold K = k0+{8..15,24..31}
    const v8bf lo = *(const v8bf*)(row + k0 + 8 * hi);
    const v8bf hi8 = *(const v8bf*)(row + k0 + 16 + 8 * hi);
    return __builtin_shufflevector(lo, hi8, 0, 1, 2, 3, 4, 5, 6, 7, 8, 9, 10, 11, 12, 13, 14, 15);
}
__device__ inline v16bf load_bfrag(const bf16* __restrict__ row, int k0, int hi) {
    // B 32x16: lane = column N; lanes 0-15 hold K = k0+{0..7,16..23}; lanes 16-31 hold K = k0+{8..15,24..31}
    const v8bf lo = *(const v8bf*)(row + k0 + 8 * hi);
    const v8bf hi8 = *(const v8bf*)(row + k0 + 16 + 8 * hi);
    return __builtin_shufflevector(lo, hi8, 0, 1, 2, 3, 4, 5, 6, 7, 8, 9, 10, 11, 12, 13, 14, 15);
}

// ---------------- kernel 4: score GEMM  Y = P * P^T, column-scaled by inv_norm ----------------
// one wave per block; wave computes a 64(M) x 32(N) tile via 4x2 v_wmma_f32_16x16x32_bf16
__global__ __launch_bounds__(32) void k_gemm_scores(const bf16* __restrict__ P,
                                                    const float* __restrict__ invn,
                                                    float* __restrict__ Y) {
    int lane = threadIdx.x;
    int lo = lane & 15, hi = lane >> 4;
    int M0 = blockIdx.y * 64;
    int N0 = blockIdx.x * 32;
    v8f acc[4][2] = {};
    for (int k0 = 0; k0 < EE; k0 += 32) {
        v16bf afr[4], bfr[2];
#pragma unroll
        for (int mi = 0; mi < 4; mi++)
            afr[mi] = load_afrag(P + (size_t)(M0 + mi * 16 + lo) * EE, k0, hi);
#pragma unroll
        for (int ni = 0; ni < 2; ni++)
            bfr[ni] = load_bfrag(P + (size_t)(N0 + ni * 16 + lo) * EE, k0, hi);
        if (k0 + 64 < EE) {   // pull next K-tiles toward the WGP (global_prefetch_b8)
#pragma unroll
            for (int mi = 0; mi < 4; mi++)
                __builtin_prefetch(P + (size_t)(M0 + mi * 16 + lo) * EE + k0 + 64, 0, 3);
#pragma unroll
            for (int ni = 0; ni < 2; ni++)
                __builtin_prefetch(P + (size_t)(N0 + ni * 16 + lo) * EE + k0 + 64, 0, 3);
        }
#pragma unroll
        for (int mi = 0; mi < 4; mi++)
#pragma unroll
            for (int ni = 0; ni < 2; ni++)
                acc[mi][ni] = __builtin_amdgcn_wmma_f32_16x16x32_bf16(
                    false, afr[mi], false, bfr[ni], (short)0, acc[mi][ni], false, false);
    }
#pragma unroll
    for (int ni = 0; ni < 2; ni++) {
        int col = N0 + ni * 16 + lo;
        float sc = invn[col];
#pragma unroll
        for (int mi = 0; mi < 4; mi++)
#pragma unroll
            for (int r = 0; r < 8; r++) {
                int row = M0 + mi * 16 + r + 8 * hi;   // D layout: VGPR r -> M=r (+8 for hi half)
                Y[(size_t)row * LL + col] = acc[mi][ni][r] * sc;
            }
    }
}

// ---------------- kernel 5: first diagonal fuse (eye(3) conv on flattened LxL) ----------------
__global__ void k_fuse1(const float* __restrict__ Y, float* __restrict__ A1) {
    size_t t = (size_t)blockIdx.x * blockDim.x + threadIdx.x;
    const size_t total = (size_t)LL * LL;
    if (t >= total) return;
    int i = (int)(t / LL), j = (int)(t % LL);
    float v = Y[t];
    if (i >= 1 && j >= 1) v += Y[t - LL - 1];
    if (i < LL - 1 && j < LL - 1) v += Y[t + LL + 1];
    A1[t] = v;
}

// ---------------- kernel 6: second fuse in (w,h)-transposed flattening ----------------
// One workgroup per OUTPUT row q=(i,j). The three rows of the transposed matrix it
// needs are contiguous rows of A1 -> stage coalesced into LDS, apply the column
// permutation as a closed-form LDS index, write the output row contiguously.
__global__ __launch_bounds__(256) void k_fuse2(const float* __restrict__ A1, float* __restrict__ Y) {
    __shared__ float lds[3 * LL];                      // 27,648 B
    int q = blockIdx.x;                                // output row
    int i = q / hh, j = q % hh;
    int r = j * hh + i;                                // row in (w,h)-transposed flattening
#pragma unroll
    for (int d = 0; d < 3; d++) {
        int rp = r + d - 1;
        if (rp >= 0 && rp < LL) {
            const float* src = A1 + (size_t)((rp % hh) * hh + rp / hh) * LL;  // rowA(rp)
            for (int t = threadIdx.x; t < LL; t += 256) lds[d * LL + t] = src[t];
        } else {
            for (int t = threadIdx.x; t < LL; t += 256) lds[d * LL + t] = 0.f;
        }
    }
    __syncthreads();
    float* outrow = Y + (size_t)q * LL;
    for (int s = threadIdx.x; s < LL; s += 256) {
        int u = s / hh, v = s % hh;                    // s = u*hh+v ; c = v*hh+u ; colA(c)==s
        float acc = lds[LL + s];                       // d = 0 term (always valid)
        if (s != 0) {                                  // c-1 >= 0 ; row validity via zero-fill
            int idxm = (u >= 1) ? (s - hh) : ((hh - 1) * hh + v - 1);   // colA(c-1)
            acc += lds[idxm];
        }
        if (s != LL - 1) {                             // c+1 <= L-1
            int idxp = (u <= hh - 2) ? (s + hh) : (v + 1);              // colA(c+1)
            acc += lds[2 * LL + idxp];
        }
        outrow[s] = acc;
    }
}

// ---------------- kernel 7: masked, scaled, stable softmax -> bf16 weights ----------------
__global__ __launch_bounds__(256) void k_softmax(const float* __restrict__ Yf,
                                                 const float* __restrict__ mm,
                                                 bf16* __restrict__ Wb) {
    __shared__ float red[256];
    int q = blockIdx.x;
    float z[9];
    float mx = -1e30f;
#pragma unroll
    for (int it = 0; it < 9; it++) {
        int s = threadIdx.x + it * 256;
        float val = Yf[(size_t)q * LL + s] * mm[s] * 10.0f;   // mask, then softmax_scale
        z[it] = val;
        mx = fmaxf(mx, val);
    }
    red[threadIdx.x] = mx;
    __syncthreads();
    for (int s2 = 128; s2 > 0; s2 >>= 1) {
        if (threadIdx.x < s2) red[threadIdx.x] = fmaxf(red[threadIdx.x], red[threadIdx.x + s2]);
        __syncthreads();
    }
    mx = red[0];
    __syncthreads();
    float sum = 0.f;
#pragma unroll
    for (int it = 0; it < 9; it++) { z[it] = __expf(z[it] - mx); sum += z[it]; }
    red[threadIdx.x] = sum;
    __syncthreads();
    for (int s2 = 128; s2 > 0; s2 >>= 1) {
        if (threadIdx.x < s2) red[threadIdx.x] += red[threadIdx.x + s2];
        __syncthreads();
    }
    float inv = 1.f / red[0];
#pragma unroll
    for (int it = 0; it < 9; it++) {
        int s = threadIdx.x + it * 256;
        Wb[(size_t)q * LL + s] = (bf16)(z[it] * inv * mm[s]);   // re-mask after softmax
    }
}

// ---------------- kernel 8: paste GEMM  T[ab] = Wb(LxL) * Xab(LxC) ----------------
__global__ __launch_bounds__(32) void k_gemm_paste(const bf16* __restrict__ Wb,
                                                   const bf16* __restrict__ XabT,
                                                   float* __restrict__ T) {
    int lane = threadIdx.x;
    int lo = lane & 15, hi = lane >> 4;
    int M0 = blockIdx.y * 64;
    int N0 = blockIdx.x * 32;
    int ab = blockIdx.z;
    const bf16* Bbase = XabT + (size_t)ab * CC * LL;
    v8f acc[4][2] = {};
    for (int k0 = 0; k0 < LL; k0 += 32) {
        v16bf afr[4], bfr[2];
#pragma unroll
        for (int mi = 0; mi < 4; mi++)
            afr[mi] = load_afrag(Wb + (size_t)(M0 + mi * 16 + lo) * LL, k0, hi);
#pragma unroll
        for (int ni = 0; ni < 2; ni++)
            bfr[ni] = load_bfrag(Bbase + (size_t)(N0 + ni * 16 + lo) * LL, k0, hi);
        if (k0 + 64 < LL) {   // global_prefetch_b8 for next K-tiles
#pragma unroll
            for (int mi = 0; mi < 4; mi++)
                __builtin_prefetch(Wb + (size_t)(M0 + mi * 16 + lo) * LL + k0 + 64, 0, 3);
#pragma unroll
            for (int ni = 0; ni < 2; ni++)
                __builtin_prefetch(Bbase + (size_t)(N0 + ni * 16 + lo) * LL + k0 + 64, 0, 3);
        }
#pragma unroll
        for (int mi = 0; mi < 4; mi++)
#pragma unroll
            for (int ni = 0; ni < 2; ni++)
                acc[mi][ni] = __builtin_amdgcn_wmma_f32_16x16x32_bf16(
                    false, afr[mi], false, bfr[ni], (short)0, acc[mi][ni], false, false);
    }
    float* Tab = T + (size_t)ab * LL * CC;
#pragma unroll
    for (int ni = 0; ni < 2; ni++) {
        int col = N0 + ni * 16 + lo;
#pragma unroll
        for (int mi = 0; mi < 4; mi++)
#pragma unroll
            for (int r = 0; r < 8; r++) {
                int row = M0 + mi * 16 + r + 8 * hi;
                Tab[(size_t)row * CC + col] = acc[mi][ni][r];
            }
    }
}

// ---------------- kernel 9: conv-transpose overlap-add (vjp of strided SAME conv), /4 ----------------
__global__ void k_gather(const float* __restrict__ T, float* __restrict__ out) {
    int idx = blockIdx.x * blockDim.x + threadIdx.x;   // HH*HH*CC
    if (idx >= HH * HH * CC) return;
    int c = idx & (CC - 1);
    int q = (idx >> 7) % HH;
    int p = idx / (HH * CC);
    float acc = 0.f;
    for (int a = (p & 1); a < 3; a += 2) {
        int hp = (p - a) >> 1;
        if (hp < 0 || hp >= hh) continue;
        for (int b = (q & 1); b < 3; b += 2) {
            int wp = (q - b) >> 1;
            if (wp < 0 || wp >= hh) continue;
            acc += T[((size_t)(a * 3 + b) * LL + hp * hh + wp) * CC + c];
        }
    }
    out[idx] = 0.25f * acc;
}

// ---------------- host: orchestrate per-sample pipeline on `stream` ----------------
extern "C" void kernel_launch(void* const* d_in, const int* in_sizes, int n_in,
                              void* d_out, int out_size, void* d_ws, size_t ws_size,
                              hipStream_t stream) {
    const float* x    = (const float*)d_in[0];
    const float* mask = (const float*)d_in[1];
    float* out = (float*)d_out;
    char* ws = (char*)d_ws;

    // per-sample workspace layout (reused across samples), all 16B-aligned
    const size_t off_P    = 0;                                   // L*E bf16   = 5,308,416
    const size_t off_invn = off_P    + (size_t)LL * EE * 2;      // L f32
    const size_t off_mm   = off_invn + (size_t)LL * 4;           // L f32
    const size_t off_f    = off_mm   + (size_t)LL * 4;           // h*h*C f32  = 1,179,648
    const size_t off_m    = off_f    + (size_t)hh * hh * CC * 4; // h*h f32
    const size_t off_Xab  = off_m    + (size_t)hh * hh * 4;      // 9*C*L bf16 = 5,308,416
    const size_t off_Y    = off_Xab  + (size_t)9 * CC * LL * 2;  // L*L f32    = 21,233,664
    const size_t off_A1   = off_Y    + (size_t)LL * LL * 4;      // L*L f32    = 21,233,664
    const size_t off_Wb   = off_A1;                              // aliases A1 (dead): L*L bf16
    const size_t off_T    = off_A1 + (size_t)LL * LL * 2;        // aliases A1 2nd half: 9*L*C f32

    bf16*  P    = (bf16*)(ws + off_P);
    float* invn = (float*)(ws + off_invn);
    float* mmv  = (float*)(ws + off_mm);
    float* fbuf = (float*)(ws + off_f);
    float* mbuf = (float*)(ws + off_m);
    bf16*  Xab  = (bf16*)(ws + off_Xab);
    float* Y    = (float*)(ws + off_Y);
    float* A1   = (float*)(ws + off_A1);
    bf16*  Wb   = (bf16*)(ws + off_Wb);
    float* T    = (float*)(ws + off_T);

    const size_t totLL = (size_t)LL * LL;

    for (int b = 0; b < BATCH; b++) {
        const float* xb = x + (size_t)b * HH * HH * CC;
        const float* mb = mask + (size_t)b * HH * HH;
        float* ob = out + (size_t)b * HH * HH * CC;

        k_pool<<<(hh * hh * CC + 255) / 256, 256, 0, stream>>>(xb, mb, fbuf, mbuf);
        k_patch<<<LL, 256, 0, stream>>>(fbuf, mbuf, P, invn, mmv);
        k_xabt<<<(9 * CC * LL + 255) / 256, 256, 0, stream>>>(xb, Xab);

        dim3 gs(LL / 32, LL / 64);               // (72, 36) one-wave blocks
        k_gemm_scores<<<gs, 32, 0, stream>>>(P, invn, Y);

        k_fuse1<<<(unsigned)((totLL + 255) / 256), 256, 0, stream>>>(Y, A1);
        k_fuse2<<<LL, 256, 0, stream>>>(A1, Y);
        k_softmax<<<LL, 256, 0, stream>>>(Y, mmv, Wb);

        dim3 gp(CC / 32, LL / 64, 9);            // (4, 36, 9) one-wave blocks
        k_gemm_paste<<<gp, 32, 0, stream>>>(Wb, Xab, T);

        k_gather<<<(HH * HH * CC + 255) / 256, 256, 0, stream>>>(T, ob);
    }
}
